// Llama3Attention_40450001994281
// MI455X (gfx1250) — matrix operand order repaired and checked by reading, not compiled, not run on hardware
//
#include <hip/hip_runtime.h>
#include <hip/hip_bf16.h>

typedef _Float16 v8h  __attribute__((ext_vector_type(8)));
typedef _Float16 v16h __attribute__((ext_vector_type(16)));
typedef float    v8f  __attribute__((ext_vector_type(8)));

#define S_LEN  2048
#define HDIM   4096
#define QKVD   6144
#define NHEADS 32
#define NKVH   8
#define HEADD  128

// ---- WMMA operand loaders (layouts per CDNA5 ISA §7.12.2) ----
// A 16x32 f16: lane&15 = row M; lanes<16 hold K {0..7, 16..23}, lanes>=16 hold K {8..15, 24..31}
__device__ __forceinline__ v16h ld_a(const _Float16* p, int lane) {
  const int off = (lane & 16) ? 8 : 0;
  v8h lo = *(const v8h*)(p + off);
  v8h hi = *(const v8h*)(p + off + 16);
  return __builtin_shufflevector(lo, hi, 0,1,2,3,4,5,6,7,8,9,10,11,12,13,14,15);
}
// B 32x16 f16: lane&15 = column N; lanes<16 hold K {0..7, 16..23}, lanes>=16 hold K {8..15, 24..31}
__device__ __forceinline__ v16h ld_b(const _Float16* p, int lane) {
  const int off = (lane & 16) ? 8 : 0;
  v8h lo = *(const v8h*)(p + off);
  v8h hi = *(const v8h*)(p + off + 16);
  return __builtin_shufflevector(lo, hi, 0,1,2,3,4,5,6,7,8,9,10,11,12,13,14,15);
}
__device__ __forceinline__ v8f wmma32(v16h a, v16h b, v8f c) {
  return __builtin_amdgcn_wmma_f32_16x16x32_f16(false, a, false, b, (short)0, c, false, false);
}

// 16-byte async copy global -> LDS (CDNA5 async engine, tracked by ASYNCcnt)
__device__ __forceinline__ void async_cp16(uint32_t lds_addr, const _Float16* gsrc) {
  asm volatile("global_load_async_to_lds_b128 %0, %1, off"
               :: "v"(lds_addr), "v"(gsrc) : "memory");
}

// ---------------- elementwise f32 -> f16 ----------------
__global__ void f2h_kernel(const float* __restrict__ in, _Float16* __restrict__ out, int n) {
  for (int i = blockIdx.x * blockDim.x + threadIdx.x; i < n; i += gridDim.x * blockDim.x)
    out[i] = (_Float16)in[i];
}

// ---------------- LDS-tiled transpose, f32 in [R][C] -> f16 out [C][R] ----------------
__global__ void transpose_f2h_kernel(const float* __restrict__ in, _Float16* __restrict__ out,
                                     int R, int C) {
  __shared__ float tile[32][33];
  const int c0 = blockIdx.x * 32;
  const int r0 = blockIdx.y * 32;
  #pragma unroll
  for (int j = 0; j < 32; j += 8)
    tile[threadIdx.y + j][threadIdx.x] =
        in[(size_t)(r0 + threadIdx.y + j) * C + c0 + threadIdx.x];
  __syncthreads();
  #pragma unroll
  for (int j = 0; j < 32; j += 8)
    out[(size_t)(c0 + threadIdx.y + j) * R + r0 + threadIdx.x] =
        (_Float16)tile[threadIdx.x][threadIdx.y + j];
}

// ---------------- WMMA GEMM: C[M,N] = A[M,K] * Bt[N,K]^T ----------------
// 4 waves / block -> 64x64 tile. B panel double-buffered in LDS via async copies;
// each wave owns 16 rows of A (kept in registers via direct global loads).
__global__ void gemm_f16_64x64(const _Float16* __restrict__ A, const _Float16* __restrict__ Bt,
                               float* __restrict__ C, int K, int N) {
  __shared__ __align__(128) _Float16 tileB[2][64 * 32];   // 2 x 4KB
  const int tid  = threadIdx.x;       // 0..127
  const int lane = tid & 31;
  const int wave = tid >> 5;          // 0..3
  const int m0 = blockIdx.x * 64;
  const int n0 = blockIdx.y * 64;

  const _Float16* arow = A + (size_t)(m0 + wave * 16 + (lane & 15)) * K;

  // stage one 64(n) x 32(k) f16 B tile: 256 x 16B chunks, 2 per thread
  auto stage = [&](int buf, int kc) {
    #pragma unroll
    for (int i = 0; i < 2; ++i) {
      const int c    = tid * 2 + i;       // 0..255
      const int r    = c >> 2;            // n-local row
      const int part = c & 3;             // 8-half slice within row
      const _Float16* src = Bt + (size_t)(n0 + r) * K + kc + part * 8;
      const uint32_t dst = (uint32_t)(size_t)&tileB[buf][c * 8];
      async_cp16(dst, src);
    }
  };

  stage(0, 0);
  v8f acc0 = {}, acc1 = {}, acc2 = {}, acc3 = {};
  for (int kc = 0; kc < K; kc += 32) {
    const int cur = (kc >> 5) & 1;
    if (kc + 32 < K) {
      stage(cur ^ 1, kc + 32);
      asm volatile("s_wait_asynccnt 2" ::: "memory");   // current tile landed
    } else {
      asm volatile("s_wait_asynccnt 0" ::: "memory");
    }
    __syncthreads();
    __builtin_prefetch(arow + kc + 256, 0, 3);
    v16h a = ld_a(arow + kc, lane);
    const _Float16* bb = &tileB[cur][(lane & 15) * 32];
    acc0 = wmma32(a, ld_b(bb + 0 * 16 * 32, lane), acc0);
    acc1 = wmma32(a, ld_b(bb + 1 * 16 * 32, lane), acc1);
    acc2 = wmma32(a, ld_b(bb + 2 * 16 * 32, lane), acc2);
    acc3 = wmma32(a, ld_b(bb + 3 * 16 * 32, lane), acc3);
    __syncthreads();
  }
  const int col   = n0 + (lane & 15);
  const int rbase = m0 + wave * 16 + ((lane & 16) ? 8 : 0);
  #pragma unroll
  for (int r = 0; r < 8; ++r) {
    const size_t ro = (size_t)(rbase + r) * N;
    C[ro + col]      = acc0[r];
    C[ro + col + 16] = acc1[r];
    C[ro + col + 32] = acc2[r];
    C[ro + col + 48] = acc3[r];
  }
}

// ---------------- RoPE + repack: qkv f32 [S][6144] -> qh[NH][S][128], kh[NKV][S][128], vT[NKV][128][S] ----------------
__global__ void rope_pack_kernel(const float* __restrict__ qkv, const float* __restrict__ rope,
                                 _Float16* __restrict__ qh, _Float16* __restrict__ kh,
                                 _Float16* __restrict__ vT) {
  const int s = blockIdx.x;
  const int j = blockIdx.y;   // 0..47 : 32 q heads, 8 k heads, 8 v heads
  const int t = threadIdx.x;  // 0..63 : pair index within head dim
  const float* row = qkv + (size_t)s * QKVD + (size_t)j * HEADD;
  if (j < NHEADS + NKVH) {
    const float c  = rope[s * HEADD + 2 * t];
    const float sn = rope[s * HEADD + 2 * t + 1];
    const float t0 = row[2 * t], t1 = row[2 * t + 1];
    const float o0 = t0 * c - t1 * sn;
    const float o1 = t1 * c + t0 * sn;
    if (j < NHEADS) {
      const float sc = 0.08838834764831845f; // 1/sqrt(HEADD), folded into q
      _Float16* dst = qh + ((size_t)j * S_LEN + s) * HEADD;
      dst[2 * t]     = (_Float16)(o0 * sc);
      dst[2 * t + 1] = (_Float16)(o1 * sc);
    } else {
      _Float16* dst = kh + ((size_t)(j - NHEADS) * S_LEN + s) * HEADD;
      dst[2 * t]     = (_Float16)o0;
      dst[2 * t + 1] = (_Float16)o1;
    }
  } else {
    const int vj = j - NHEADS - NKVH;
    vT[((size_t)vj * HEADD + 2 * t) * S_LEN + s]     = (_Float16)row[2 * t];
    vT[((size_t)vj * HEADD + 2 * t + 1) * S_LEN + s] = (_Float16)row[2 * t + 1];
  }
}

// ---------------- Flash attention: one wave per (16 query rows, head) ----------------
__global__ void flash_attn_kernel(const _Float16* __restrict__ qh, const _Float16* __restrict__ kh,
                                  const _Float16* __restrict__ vT, _Float16* __restrict__ aout) {
  __shared__ __align__(64) _Float16 pbuf[16 * 32];
  const int lane   = threadIdx.x;
  const int qt     = blockIdx.x;
  const int h      = blockIdx.y;
  const int kvh    = h >> 2;                 // GQA: 4 q heads per kv head
  const int qbase  = qt * 16;
  const int lrow   = lane & 15;
  const int rowoff = (lane & 16) ? 8 : 0;

  const _Float16* qrow = qh + ((size_t)h * S_LEN + qbase + lrow) * HEADD;
  v16h qa0 = ld_a(qrow +  0, lane);
  v16h qa1 = ld_a(qrow + 32, lane);
  v16h qa2 = ld_a(qrow + 64, lane);
  v16h qa3 = ld_a(qrow + 96, lane);

  v8f o[8];
  #pragma unroll
  for (int i = 0; i < 8; ++i) o[i] = (v8f){};
  float mprev[8], lsum[8];
  #pragma unroll
  for (int r = 0; r < 8; ++r) { mprev[r] = -__builtin_inff(); lsum[r] = 0.0f; }

  const int qmax = qbase + 15;
  for (int kb = 0; kb <= qmax; kb += 32) {
    const _Float16* krow0 = kh + ((size_t)kvh * S_LEN + kb + lrow) * HEADD;
    const _Float16* krow1 = krow0 + 16 * HEADD;
    v8f s0 = {}, s1 = {};
    s0 = wmma32(qa0, ld_b(krow0 +  0, lane), s0);
    s1 = wmma32(qa0, ld_b(krow1 +  0, lane), s1);
    s0 = wmma32(qa1, ld_b(krow0 + 32, lane), s0);
    s1 = wmma32(qa1, ld_b(krow1 + 32, lane), s1);
    s0 = wmma32(qa2, ld_b(krow0 + 64, lane), s0);
    s1 = wmma32(qa2, ld_b(krow1 + 64, lane), s1);
    s0 = wmma32(qa3, ld_b(krow0 + 96, lane), s0);
    s1 = wmma32(qa3, ld_b(krow1 + 96, lane), s1);

    // causal mask + online softmax in C-register layout (row M = r + rowoff, col = lane&15)
    #pragma unroll
    for (int r = 0; r < 8; ++r) {
      const int qg = qbase + rowoff + r;
      float a0 = (kb + lrow <= qg)      ? s0[r] : -__builtin_inff();
      float a1 = (kb + 16 + lrow <= qg) ? s1[r] : -__builtin_inff();
      float mx = fmaxf(a0, a1);
      mx = fmaxf(mx, __shfl_xor(mx, 1, 32));
      mx = fmaxf(mx, __shfl_xor(mx, 2, 32));
      mx = fmaxf(mx, __shfl_xor(mx, 4, 32));
      mx = fmaxf(mx, __shfl_xor(mx, 8, 32));
      const float mnew  = fmaxf(mprev[r], mx);
      const float scale = __expf(mprev[r] - mnew);
      const float e0 = __expf(a0 - mnew);
      const float e1 = __expf(a1 - mnew);
      float sum = e0 + e1;
      sum += __shfl_xor(sum, 1, 32);
      sum += __shfl_xor(sum, 2, 32);
      sum += __shfl_xor(sum, 4, 32);
      sum += __shfl_xor(sum, 8, 32);
      lsum[r]  = lsum[r] * scale + sum;
      mprev[r] = mnew;
      #pragma unroll
      for (int dc = 0; dc < 8; ++dc) o[dc][r] *= scale;
      pbuf[(rowoff + r) * 32 + lrow]      = (_Float16)e0;
      pbuf[(rowoff + r) * 32 + 16 + lrow] = (_Float16)e1;
    }
    __syncthreads();
    v16h pa = ld_a(pbuf + lrow * 32, lane);   // re-enter P in A-operand layout
    __syncthreads();

    // O += P (16x32) * V (32 keys x 128 dims); vT rows are [d][s] so B tiles are contiguous
    const _Float16* vb = vT + ((size_t)kvh * HEADD + lrow) * S_LEN + kb;
    #pragma unroll
    for (int dc = 0; dc < 8; ++dc)
      o[dc] = wmma32(pa, ld_b(vb + (size_t)dc * 16 * S_LEN, lane), o[dc]);
  }

  // normalize and write attn output row-major [S][NH*HD]
  #pragma unroll
  for (int r = 0; r < 8; ++r) {
    const float inv = 1.0f / lsum[r];
    const size_t base = (size_t)(qbase + rowoff + r) * HDIM + (size_t)h * HEADD;
    #pragma unroll
    for (int dc = 0; dc < 8; ++dc)
      aout[base + dc * 16 + lrow] = (_Float16)(o[dc][r] * inv);
  }
}

extern "C" void kernel_launch(void* const* d_in, const int* in_sizes, int n_in,
                              void* d_out, int out_size, void* d_ws, size_t ws_size,
                              hipStream_t stream) {
  (void)in_sizes; (void)n_in; (void)out_size; (void)ws_size;
  const float* x    = (const float*)d_in[0];
  // d_in[1] = last_pos (zeros), d_in[2] = mask (causal, recomputed analytically)
  const float* rope = (const float*)d_in[3];
  const float* wqkv = (const float*)d_in[4];
  const float* wo   = (const float*)d_in[5];
  float* out = (float*)d_out;

  char* ws = (char*)d_ws;
  size_t cur = 0;
  auto alloc = [&](size_t bytes) -> void* {
    void* p = ws + cur;
    cur += (bytes + 255) & ~(size_t)255;
    return p;
  };
  _Float16* xh    = (_Float16*)alloc((size_t)S_LEN * HDIM * 2);
  _Float16* wqkvT = (_Float16*)alloc((size_t)QKVD * HDIM * 2);
  _Float16* woT   = (_Float16*)alloc((size_t)HDIM * HDIM * 2);
  float*    qkv   = (float*)   alloc((size_t)S_LEN * QKVD * 4);
  _Float16* qhb   = (_Float16*)alloc((size_t)NHEADS * S_LEN * HEADD * 2);
  _Float16* khb   = (_Float16*)alloc((size_t)NKVH * S_LEN * HEADD * 2);
  _Float16* vTb   = (_Float16*)alloc((size_t)NKVH * HEADD * S_LEN * 2);
  _Float16* attnh = (_Float16*)alloc((size_t)S_LEN * HDIM * 2);

  f2h_kernel<<<4096, 256, 0, stream>>>(x, xh, S_LEN * HDIM);
  transpose_f2h_kernel<<<dim3(QKVD / 32, HDIM / 32), dim3(32, 8), 0, stream>>>(wqkv, wqkvT, HDIM, QKVD);
  transpose_f2h_kernel<<<dim3(HDIM / 32, HDIM / 32), dim3(32, 8), 0, stream>>>(wo, woT, HDIM, HDIM);
  gemm_f16_64x64<<<dim3(S_LEN / 64, QKVD / 64), 128, 0, stream>>>(xh, wqkvT, qkv, HDIM, QKVD);
  rope_pack_kernel<<<dim3(S_LEN, NHEADS + 2 * NKVH), 64, 0, stream>>>(qkv, rope, qhb, khb, vTb);
  flash_attn_kernel<<<dim3(S_LEN / 16, NHEADS), 32, 0, stream>>>(qhb, khb, vTb, attnh);
  gemm_f16_64x64<<<dim3(S_LEN / 64, HDIM / 64), 128, 0, stream>>>(attnh, woT, out, HDIM, HDIM);
}
